// SingleAttention_61375082660168
// MI455X (gfx1250) — matrix operand order repaired and checked by reading, not compiled, not run on hardware
//
#include <hip/hip_runtime.h>

typedef __attribute__((ext_vector_type(16))) _Float16 v16h;
typedef __attribute__((ext_vector_type(8)))  _Float16 v8h;
typedef __attribute__((ext_vector_type(4)))  _Float16 v4h;
typedef __attribute__((ext_vector_type(8)))  float    v8f;
typedef __attribute__((ext_vector_type(4)))  float    v4f;

#define BB 8
#define SS 4096
#define DIN 128
#define DK 64
#define KT 128                       // keys per inner tile

// scale folded into Q at projection time: (1/sqrt(64)) * log2(e)
#define QSCALE 0.18033688011112042f

// Concatenate two contiguous 8-f16 runs into one A-fragment register set.
__device__ __forceinline__ v16h cat8(v8h lo, v8h hi) {
    return __builtin_shufflevector(lo, hi, 0, 1, 2, 3, 4, 5, 6, 7,
                                   8, 9, 10, 11, 12, 13, 14, 15);
}

// ---------------------------------------------------------------------------
// Kernel 1: fused Q/K/V projection.  grid = (B*S/128, 3), block = 256 (8 waves).
// W staged TRANSPOSED in LDS -> B-fragments are contiguous v16h loads.
// Q is pre-scaled by 0.125*log2(e) so attention softmax runs in base-2 with
// no per-score multiplies.  V stored TRANSPOSED (Vt[b][dim][s]) so attention
// V B-fragments are contiguous 32-byte global loads.
// ---------------------------------------------------------------------------
__global__ __launch_bounds__(256) void proj_qkv(
    const float* __restrict__ q_in, const float* __restrict__ k_in,
    const float* __restrict__ v_in,
    const float* __restrict__ Wq, const float* __restrict__ bq,
    const float* __restrict__ Wk, const float* __restrict__ bk,
    const float* __restrict__ Wv, const float* __restrict__ bv,
    _Float16* __restrict__ Qh, _Float16* __restrict__ Kh,
    _Float16* __restrict__ Vt)
{
    __shared__ _Float16 sA[128 * DIN];    // 32 KB: 128 input rows (f16)
    __shared__ _Float16 sWt[DK * DIN];    // 16 KB: W transposed [n][k]
    __shared__ float    sBias[DK];

    const int tid = threadIdx.x;
    const int sel = blockIdx.y;
    const float* x    = (sel == 0) ? q_in : (sel == 1) ? k_in : v_in;
    const float* W    = (sel == 0) ? Wq   : (sel == 1) ? Wk   : Wv;
    const float* bias = (sel == 0) ? bq   : (sel == 1) ? bk   : bv;
    _Float16* outp    = (sel == 0) ? Qh   : (sel == 1) ? Kh   : Vt;
    const float oscale = (sel == 0) ? QSCALE : 1.0f;

    const long row0 = (long)blockIdx.x * 128;

    // stage inputs as f16 (vectorized f32x4 -> f16x4)
    for (int i = tid * 4; i < 128 * DIN; i += 1024) {
        v4f f = *(const v4f*)&x[row0 * DIN + i];
        v4h hh;
        hh[0] = (_Float16)f[0]; hh[1] = (_Float16)f[1];
        hh[2] = (_Float16)f[2]; hh[3] = (_Float16)f[3];
        *(v4h*)&sA[i] = hh;
    }
    // stage W transposed
    for (int i = tid * 4; i < DIN * DK; i += 1024) {
        const int k = i / DK, n = i % DK;
        v4f f = *(const v4f*)&W[i];
#pragma unroll
        for (int t = 0; t < 4; ++t) sWt[(n + t) * DIN + k] = (_Float16)f[t];
    }
    if (tid < DK) sBias[tid] = bias[tid];
    __syncthreads();

    const int wave = tid >> 5;
    const int lane = tid & 31;
    const int h    = lane >> 4;   // lane half
    const int nrow = lane & 15;   // A-row within slab / B,C,D column

    // A fragments: two contiguous 8-f16 runs per 16x32 chunk
    v16h afrag[4];
#pragma unroll
    for (int c = 0; c < 4; ++c) {
        const int off = (wave * 16 + nrow) * DIN + 32 * c + 16 * h;
        afrag[c] = cat8(*(const v8h*)&sA[off], *(const v8h*)&sA[off + 8]);
    }

#pragma unroll
    for (int nt = 0; nt < 4; ++nt) {
        v8f acc = {};
#pragma unroll
        for (int c = 0; c < 4; ++c) {
            v16h bfrag = *(const v16h*)&sWt[(nt * 16 + nrow) * DIN + 32 * c + 16 * h];
            acc = __builtin_amdgcn_wmma_f32_16x16x32_f16(
                false, afrag[c], false, bfrag, (short)0, acc, false, false);
        }
        const float bval = sBias[nt * 16 + nrow];
        if (sel == 2) {
            // transposed store: Vt[(b*64 + dim) * S + s]
            const long bIdx = row0 / SS;
#pragma unroll
            for (int r = 0; r < 8; ++r) {
                const long row = row0 + wave * 16 + r + 8 * h;
                const long s   = row - bIdx * SS;
                outp[(bIdx * DK + nt * 16 + nrow) * (long)SS + s] =
                    (_Float16)(acc[r] + bval);
            }
        } else {
#pragma unroll
            for (int r = 0; r < 8; ++r) {
                const long row = row0 + wave * 16 + r + 8 * h;
                outp[row * DK + nt * 16 + nrow] =
                    (_Float16)((acc[r] + bval) * oscale);
            }
        }
    }
}

// ---------------------------------------------------------------------------
// Kernel 2: flash attention. grid = (S/128, B), block = 256 (8 waves).
// 128-key inner tiles amortize per-tile softmax bookkeeping (max tree, alpha
// rescale, l update) over 2x the keys.  K/V B-fragments are contiguous
// 32-byte global loads (L2-resident, broadcast).  Base-2 softmax; row sums
// on the matrix pipe via P @ ones.
// ---------------------------------------------------------------------------
__global__ __launch_bounds__(256) void flash_attn(
    const _Float16* __restrict__ Qh, const _Float16* __restrict__ Kh,
    const _Float16* __restrict__ Vt, float* __restrict__ out)
{
    __shared__ _Float16 sP[8][16 * KT];   // 32 KB per-wave P scratch

    const int tid  = threadIdx.x;
    const int wave = tid >> 5;
    const int lane = tid & 31;
    const int h    = lane >> 4;
    const int nrow = lane & 15;

    const int  b    = blockIdx.y;
    const long base = (long)b * SS;                    // batch row offset
    const int  q0   = blockIdx.x * 128 + wave * 16;    // q row within batch

    // Q fragments (16 rows x 64 dims -> two 16x32 A fragments), resident
    v16h qf[2];
#pragma unroll
    for (int c = 0; c < 2; ++c) {
        const _Float16* qp = Qh + (base + q0 + nrow) * DK + 32 * c + 16 * h;
        qf[c] = cat8(*(const v8h*)qp, *(const v8h*)(qp + 8));
    }

    // all-ones B fragment for row-sum WMMA
    v16h ones;
#pragma unroll
    for (int j = 0; j < 16; ++j) ones[j] = (_Float16)1.0f;

    float mrow[8], lrow[8];
    v8f oacc[4];
#pragma unroll
    for (int r = 0; r < 8; ++r) { mrow[r] = -1e30f; lrow[r] = 0.f; }
#pragma unroll
    for (int nt = 0; nt < 4; ++nt) { v8f z = {}; oacc[nt] = z; }

    for (int kt = 0; kt < SS / KT; ++kt) {
        // prefetch next K/V tile lines this lane will touch
        if (kt + 1 < SS / KT) {
            __builtin_prefetch(Kh + (base + (long)(kt + 1) * KT + nrow) * DK, 0, 1);
            __builtin_prefetch(Vt + ((long)b * DK + nrow) * SS + (kt + 1) * KT, 0, 1);
        }

        // ---- scores S = Q' K^T (scale pre-folded into Q): 8 n-tiles ----
        v8f sc[8];
#pragma unroll
        for (int nt = 0; nt < 8; ++nt) {
            v8f acc = {};
#pragma unroll
            for (int c = 0; c < 2; ++c) {
                // B[k][n] = K[key=n][dim=k]; contiguous dims per lane row
                v16h bf = *(const v16h*)(Kh +
                    (base + (long)kt * KT + nt * 16 + nrow) * DK + 32 * c + 16 * h);
                acc = __builtin_amdgcn_wmma_f32_16x16x32_f16(
                    false, qf[c], false, bf, (short)0, acc, false, false);
            }
            sc[nt] = acc;
        }

        // ---- online softmax (base 2): row max via 16-lane shuffle tree ----
        float tmax[8];
#pragma unroll
        for (int r = 0; r < 8; ++r) {
            float t = sc[0][r];
#pragma unroll
            for (int nt = 1; nt < 8; ++nt) t = fmaxf(t, sc[nt][r]);
#pragma unroll
            for (int off = 1; off < 16; off <<= 1)
                t = fmaxf(t, __shfl_xor(t, off, 32));
            tmax[r] = t;
        }

        float alpha[8], mnew[8];
#pragma unroll
        for (int r = 0; r < 8; ++r) {
            mnew[r]  = fmaxf(mrow[r], tmax[r]);
            alpha[r] = exp2f(mrow[r] - mnew[r]);
            mrow[r]  = mnew[r];
        }

        // p = 2^(s - mnew); write C-layout P into row-major LDS
#pragma unroll
        for (int nt = 0; nt < 8; ++nt)
#pragma unroll
            for (int r = 0; r < 8; ++r) {
                const float p = exp2f(sc[nt][r] - mnew[r]);
                sP[wave][(r + 8 * h) * KT + nt * 16 + nrow] = (_Float16)p;
            }

        // rescale output accumulators while the P stores drain
#pragma unroll
        for (int nt = 0; nt < 4; ++nt)
#pragma unroll
            for (int r = 0; r < 8; ++r) oacc[nt][r] *= alpha[r];

        __builtin_amdgcn_wave_barrier();   // LDS is in-order per wave

        // reload P as A fragments (four 16x32 chunks)
        v16h pf[4];
#pragma unroll
        for (int c = 0; c < 4; ++c) {
            const _Float16* pp = &sP[wave][nrow * KT + 32 * c + 16 * h];
            pf[c] = cat8(*(const v8h*)pp, *(const v8h*)(pp + 8));
        }

        // ---- O += P(16x128) @ V(128x64) from transposed global V ----
#pragma unroll
        for (int nt = 0; nt < 4; ++nt)
#pragma unroll
            for (int c = 0; c < 4; ++c) {
                // B[k][n] = V[key=k][dim=n] = Vt[n][key]; contiguous keys
                v16h bf = *(const v16h*)(Vt +
                    ((long)b * DK + nt * 16 + nrow) * SS + (long)kt * KT + 32 * c + 16 * h);
                oacc[nt] = __builtin_amdgcn_wmma_f32_16x16x32_f16(
                    false, pf[c], false, bf, (short)0, oacc[nt], false, false);
            }

        // ---- row sums on the matrix pipe: rsum = P @ ones ----
        v8f racc = {};
#pragma unroll
        for (int c = 0; c < 4; ++c)
            racc = __builtin_amdgcn_wmma_f32_16x16x32_f16(
                false, pf[c], false, ones, (short)0, racc, false, false);
#pragma unroll
        for (int r = 0; r < 8; ++r)
            lrow[r] = lrow[r] * alpha[r] + racc[r];
    }

    // ---- normalize and store f32 ----
#pragma unroll
    for (int r = 0; r < 8; ++r) {
        const float inv = 1.0f / lrow[r];
        const long  row = base + q0 + r + 8 * h;
#pragma unroll
        for (int nt = 0; nt < 4; ++nt)
            out[row * DK + nt * 16 + nrow] = oacc[nt][r] * inv;
    }
}

// ---------------------------------------------------------------------------
extern "C" void kernel_launch(void* const* d_in, const int* in_sizes, int n_in,
                              void* d_out, int out_size, void* d_ws, size_t ws_size,
                              hipStream_t stream) {
    const float* q_in = (const float*)d_in[0];
    const float* k_in = (const float*)d_in[1];
    const float* v_in = (const float*)d_in[2];
    const float* Wq   = (const float*)d_in[3];
    const float* bq   = (const float*)d_in[4];
    const float* Wk   = (const float*)d_in[5];
    const float* bk   = (const float*)d_in[6];
    const float* Wv   = (const float*)d_in[7];
    const float* bv   = (const float*)d_in[8];

    const size_t nrows = (size_t)BB * SS;           // 32768
    _Float16* Qh = (_Float16*)d_ws;
    _Float16* Kh = Qh + nrows * DK;
    _Float16* Vt = Kh + nrows * DK;                 // [B][64][S] transposed

    dim3 g1((unsigned)(nrows / 128), 3);
    proj_qkv<<<g1, 256, 0, stream>>>(q_in, k_in, v_in, Wq, bq, Wk, bk, Wv, bv,
                                     Qh, Kh, Vt);

    dim3 g2(SS / 128, BB);
    flash_attn<<<g2, 256, 0, stream>>>(Qh, Kh, Vt, (float*)d_out);
}
